// SparseSequentialMLP_77343771066811
// MI455X (gfx1250) — matrix operand order repaired and checked by reading, not compiled, not run on hardware
//
#include <hip/hip_runtime.h>
#include <hip/hip_bf16.h>

// ---------------------------------------------------------------------------
// SparseSequentialMLP for MI455X (gfx1250, wave32, WMMA)
//
//   h = relu(x @ (Wi*mi)^T + bi)  x4, then  out = h @ W5^T + b5
//
// Strategy: masks are unstructured (10% density) -> fold into bf16-packed
// weights once per launch, then dense v_wmma_f32_16x16x32_bf16 GEMMs.
// Fragments are loaded straight from global memory (both operands are
// K-contiguous row-major, matching the ISA 16-bit A/B VGPR layouts with
// aligned 128-bit loads). One wave owns a 64x64 C tile = 16 WMMA tiles.
// ---------------------------------------------------------------------------

typedef __attribute__((ext_vector_type(16))) __bf16        bf16x16;
typedef __attribute__((ext_vector_type(8)))  __bf16        bf16x8;
typedef __attribute__((ext_vector_type(8)))  float         f32x8;
typedef __attribute__((ext_vector_type(8)))  unsigned short u16x8;

#define BATCH   16384
#define HIDDEN  2048
#define IN_DIM  784
#define IN_PAD  800          // next multiple of 32
#define OUT_DIM 10

__device__ __forceinline__ unsigned short f2bf_rne(float f) {
  unsigned int u = __float_as_uint(f);
  unsigned int r = u + 0x7FFFu + ((u >> 16) & 1u);   // round-to-nearest-even
  return (unsigned short)(r >> 16);
}
__device__ __forceinline__ float bf2f(unsigned short s) {
  return __uint_as_float(((unsigned int)s) << 16);
}

// --- pack x (16384,784) f32 -> (16384,800) bf16, zero padded ---------------
__global__ void pack_x_kernel(const float* __restrict__ x,
                              unsigned short* __restrict__ xb) {
  long long idx = (long long)blockIdx.x * blockDim.x + threadIdx.x;
  const long long total = (long long)BATCH * IN_PAD;
  if (idx >= total) return;
  int row = (int)(idx / IN_PAD);
  int k   = (int)(idx - (long long)row * IN_PAD);
  float v = (k < IN_DIM) ? x[(long long)row * IN_DIM + k] : 0.0f;
  xb[idx] = f2bf_rne(v);
}

// --- pack masked weights (N,K) f32 * bool -> (N,Kpad) bf16 -----------------
__global__ void pack_w_kernel(const float* __restrict__ W,
                              const unsigned char* __restrict__ m, // jnp.bool_: 1 byte
                              unsigned short* __restrict__ dst,
                              int K, int Kpad, long long total) {
  long long idx = (long long)blockIdx.x * blockDim.x + threadIdx.x;
  if (idx >= total) return;
  int row = (int)(idx / Kpad);
  int k   = (int)(idx - (long long)row * Kpad);
  float v = 0.0f;
  if (k < K) {
    long long s = (long long)row * K + k;
    v = m[s] ? W[s] : 0.0f;
  }
  dst[idx] = f2bf_rne(v);
}

// --- bf16 WMMA GEMM:  H = relu(A @ B^T + bias),  A:(M,Kpad)  B:(N,Kpad) ----
// block = 256 threads = 8 waves (2x4), wave tile = 64x64, block tile 128x256
__global__ __launch_bounds__(256)
void gemm_bias_relu_bf16(const __bf16* __restrict__ A,
                         const __bf16* __restrict__ B,
                         const float*  __restrict__ bias,
                         unsigned short* __restrict__ H,   // (M, Nout) bf16 bits
                         int Kpad, int Nout) {
  const int tid  = threadIdx.x;
  const int lane = tid & 31;
  const int wave = tid >> 5;
  const int l15  = lane & 15;
  const int lhi  = lane >> 4;                 // 0 | 1 (lane half)

  const int mBase = blockIdx.x * 128 + (wave >> 2) * 64;
  const int nBase = blockIdx.y * 256 + (wave & 3) * 64;

  // ISA 7.12.2 bf16 fragment layouts (wave32):
  //  A 16x32: lane half selects K base 0/8; halves [k0+aOff, +8) and [k0+16+aOff, +8)
  //  B 32x16: lane half selects K base 0/8; halves [k0+bOff, +8) and [k0+16+bOff, +8)
  const int aOff = lhi * 8;
  const int bOff = lhi * 8;

  const __bf16* aRow[4];
  const __bf16* bRow[4];
#pragma unroll
  for (int t = 0; t < 4; ++t) {
    aRow[t] = A + (size_t)(mBase + t * 16 + l15) * Kpad + aOff;
    bRow[t] = B + (size_t)(nBase + t * 16 + l15) * Kpad + bOff;
  }

  f32x8 acc[4][4] = {};

  for (int k0 = 0; k0 < Kpad; k0 += 32) {
    bf16x16 a[4], b[4];
#pragma unroll
    for (int t = 0; t < 4; ++t) {
      bf16x8 alo = *reinterpret_cast<const bf16x8*>(aRow[t] + k0);        // 16B aligned
      bf16x8 ahi = *reinterpret_cast<const bf16x8*>(aRow[t] + k0 + 16);
      a[t] = __builtin_shufflevector(alo, ahi, 0,1,2,3,4,5,6,7,8,9,10,11,12,13,14,15);
      bf16x8 blo = *reinterpret_cast<const bf16x8*>(bRow[t] + k0);        // 16B aligned
      bf16x8 bhi = *reinterpret_cast<const bf16x8*>(bRow[t] + k0 + 16);
      b[t] = __builtin_shufflevector(blo, bhi, 0,1,2,3,4,5,6,7,8,9,10,11,12,13,14,15);
    }

    if (k0 + 32 < Kpad) {                     // gfx1250 global_prefetch of next K tile
#pragma unroll
      for (int t = 0; t < 4; ++t) {
        __builtin_prefetch(aRow[t] + k0 + 32, 0, 3);
        __builtin_prefetch(bRow[t] + k0 + 32, 0, 3);
      }
    }

#pragma unroll
    for (int mt = 0; mt < 4; ++mt)
#pragma unroll
      for (int nt = 0; nt < 4; ++nt)
        acc[mt][nt] = __builtin_amdgcn_wmma_f32_16x16x32_bf16(
            /*neg_a=*/false, a[mt], /*neg_b=*/false, b[nt],
            /*c_mod=*/(short)0, acc[mt][nt],
            /*reuse_a=*/false, /*reuse_b=*/false);
  }

  // Epilogue. C/D f32 layout: VGPR r -> (M = tile + 8*lhi + r, N = tile + l15)
#pragma unroll
  for (int nt = 0; nt < 4; ++nt) {
    const int n   = nBase + nt * 16 + l15;
    const float bv = bias[n];
#pragma unroll
    for (int mt = 0; mt < 4; ++mt) {
      const int m0 = mBase + mt * 16 + lhi * 8;
      f32x8 c = acc[mt][nt];
#pragma unroll
      for (int r = 0; r < 8; ++r) {
        float v = c[r] + bv;
        v = v > 0.0f ? v : 0.0f;
        H[(size_t)(m0 + r) * Nout + n] = f2bf_rne(v);
      }
    }
  }
}

// --- head: out(16384,10) = h_bf16 @ W5^T + b5, f32 -------------------------
__global__ __launch_bounds__(256)
void head_kernel(const unsigned short* __restrict__ h,   // (BATCH, HIDDEN) bf16 bits
                 const float* __restrict__ W5,           // (OUT_DIM, HIDDEN)
                 const float* __restrict__ b5,
                 float* __restrict__ out) {
  int o = blockIdx.x * blockDim.x + threadIdx.x;
  if (o >= BATCH * OUT_DIM) return;
  int row = o / OUT_DIM;
  int col = o - row * OUT_DIM;
  const u16x8* hp = reinterpret_cast<const u16x8*>(h + (size_t)row * HIDDEN);
  const float* wp = W5 + (size_t)col * HIDDEN;
  float sum = b5[col];
  for (int k8 = 0; k8 < HIDDEN / 8; ++k8) {
    u16x8 hv = hp[k8];
#pragma unroll
    for (int j = 0; j < 8; ++j)
      sum += bf2f(hv[j]) * wp[k8 * 8 + j];
  }
  out[o] = sum;
}

// ---------------------------------------------------------------------------
extern "C" void kernel_launch(void* const* d_in, const int* in_sizes, int n_in,
                              void* d_out, int out_size, void* d_ws, size_t ws_size,
                              hipStream_t stream) {
  const float* x  = (const float*)d_in[0];
  const float* W1 = (const float*)d_in[1];
  const float* b1 = (const float*)d_in[2];
  const float* W2 = (const float*)d_in[3];
  const float* b2 = (const float*)d_in[4];
  const float* W3 = (const float*)d_in[5];
  const float* b3 = (const float*)d_in[6];
  const float* W4 = (const float*)d_in[7];
  const float* b4 = (const float*)d_in[8];
  const float* W5 = (const float*)d_in[9];
  const float* b5 = (const float*)d_in[10];
  const unsigned char* m1 = (const unsigned char*)d_in[11];
  const unsigned char* m2 = (const unsigned char*)d_in[12];
  const unsigned char* m3 = (const unsigned char*)d_in[13];
  const unsigned char* m4 = (const unsigned char*)d_in[14];
  (void)in_sizes; (void)n_in; (void)out_size; (void)ws_size;

  char* ws = (char*)d_ws;
  const size_t SZ_W = (size_t)HIDDEN * HIDDEN * 2;   // 8 MiB, bf16
  const size_t SZ_H = (size_t)BATCH * HIDDEN * 2;    // 64 MiB, bf16

  unsigned short* w2m = (unsigned short*)(ws);
  unsigned short* w3m = (unsigned short*)(ws + SZ_W);
  unsigned short* w4m = (unsigned short*)(ws + 2 * SZ_W);
  unsigned short* h0  = (unsigned short*)(ws + 3 * SZ_W);
  unsigned short* h1  = (unsigned short*)(ws + 3 * SZ_W + SZ_H);
  // x_bf16 and W1m are dead after layer 1; alias them into the h1 region
  // (h1 is first written by layer 2).  26.2MB + 3.3MB < 64MB.  ~152MB total.
  unsigned short* xb  = h1;
  unsigned short* w1m = (unsigned short*)((char*)h1 + (size_t)BATCH * IN_PAD * 2);

  // ---- pack inputs / masked weights to bf16 -------------------------------
  {
    long long tot = (long long)BATCH * IN_PAD;
    pack_x_kernel<<<(unsigned)((tot + 255) / 256), 256, 0, stream>>>(x, xb);
  }
  {
    long long tot = (long long)HIDDEN * IN_PAD;
    pack_w_kernel<<<(unsigned)((tot + 255) / 256), 256, 0, stream>>>(W1, m1, w1m, IN_DIM, IN_PAD, tot);
  }
  {
    long long tot = (long long)HIDDEN * HIDDEN;
    unsigned g = (unsigned)((tot + 255) / 256);
    pack_w_kernel<<<g, 256, 0, stream>>>(W2, m2, w2m, HIDDEN, HIDDEN, tot);
    pack_w_kernel<<<g, 256, 0, stream>>>(W3, m3, w3m, HIDDEN, HIDDEN, tot);
    pack_w_kernel<<<g, 256, 0, stream>>>(W4, m4, w4m, HIDDEN, HIDDEN, tot);
  }

  // ---- 4x WMMA GEMM + bias + relu (ping-pong h0/h1) -----------------------
  dim3 grid(BATCH / 128, HIDDEN / 256);   // (128, 8)
  gemm_bias_relu_bf16<<<grid, 256, 0, stream>>>(
      (const __bf16*)xb, (const __bf16*)w1m, b1, h0, IN_PAD, HIDDEN);
  gemm_bias_relu_bf16<<<grid, 256, 0, stream>>>(
      (const __bf16*)h0, (const __bf16*)w2m, b2, h1, HIDDEN, HIDDEN);
  gemm_bias_relu_bf16<<<grid, 256, 0, stream>>>(
      (const __bf16*)h1, (const __bf16*)w3m, b3, h0, HIDDEN, HIDDEN);
  gemm_bias_relu_bf16<<<grid, 256, 0, stream>>>(
      (const __bf16*)h0, (const __bf16*)w4m, b4, h1, HIDDEN, HIDDEN);

  // ---- head ---------------------------------------------------------------
  head_kernel<<<(BATCH * OUT_DIM + 255) / 256, 256, 0, stream>>>(
      h1, W5, b5, (float*)d_out);
}
